// GCNResidualBlock_59837484368079
// MI455X (gfx1250) — matrix operand order repaired and checked by reading, not compiled, not run on hardware
//
#include <hip/hip_runtime.h>

typedef _Float16 f16;
typedef __attribute__((ext_vector_type(8)))  _Float16 h8;
typedef __attribute__((ext_vector_type(16))) _Float16 v16h;
typedef __attribute__((ext_vector_type(8)))  float    v8f;

#define B_    16
#define CIN   32
#define HH    256
#define WW    256
#define COUT  32
#define TD    256
#define GH    16
#define GW    16
#define NNODE (B_*GH*GW)

__device__ __forceinline__ v16h mk16(h8 lo, h8 hi) {
  return __builtin_shufflevector(lo, hi, 0,1,2,3,4,5,6,7,8,9,10,11,12,13,14,15);
}
__device__ __forceinline__ float sigmoidf_(float x) { return 1.f / (1.f + __expf(-x)); }

// ---------------- K0: conv1 (3x3, 32->32) implicit GEMM, WMMA f16, out NHWC f16 ----
__global__ __launch_bounds__(256) void conv1_kernel(
    const float* __restrict__ x, const float* __restrict__ conv_w,
    const float* __restrict__ conv_b, f16* __restrict__ y1) {
  __shared__ __align__(16) f16 slab[3 * 132 * 32];   // [row][col(130 used)][ci]
  __shared__ __align__(16) f16 wlds[288 * 32];       // [co][k = tap*32+ci]
  const int wt = blockIdx.x, h = blockIdx.y, b = blockIdx.z;
  const int tid = threadIdx.x, w0 = wt * 128;

  for (int e = tid; e < 288 * 32; e += 256) {
    int co = e / 288, k = e % 288;
    int tap = k / 32, ci = k % 32;
    wlds[co * 288 + k] = (f16)conv_w[(co * 32 + ci) * 9 + tap];
  }
  for (int e = tid; e < 3 * 130 * 32; e += 256) {
    int col = e % 130, rc = e / 130;
    int r = rc / 32, c = rc % 32;
    int y = h + r - 1, xg = w0 + col - 1;
    float v = 0.f;
    if ((unsigned)y < 256u && (unsigned)xg < 256u)
      v = x[((b * CIN + c) * HH + y) * WW + xg];
    slab[(r * 132 + col) * 32 + c] = (f16)v;
  }
  __syncthreads();

  const int lane = tid & 31, wq = tid >> 5;
  const int ml = lane & 15, hi = lane >> 4;
  v8f acc0 = {}, acc1 = {};
#pragma unroll
  for (int tap = 0; tap < 9; ++tap) {
    int r = tap / 3, s = tap % 3;
    int col = wq * 16 + ml + s;
    const f16* ap = &slab[(r * 132 + col) * 32 + hi * 8];
    v16h A = mk16(*(const h8*)ap, *(const h8*)(ap + 16));
    const f16* bp = &wlds[ml * 288 + tap * 32 + hi * 8];
    v16h B0 = mk16(*(const h8*)bp,        *(const h8*)(bp + 16));
    v16h B1 = mk16(*(const h8*)(bp + 16 * 288), *(const h8*)(bp + 16 * 288 + 16));
    acc0 = __builtin_amdgcn_wmma_f32_16x16x32_f16(false, A, false, B0, (short)0, acc0, false, false);
    acc1 = __builtin_amdgcn_wmma_f32_16x16x32_f16(false, A, false, B1, (short)0, acc1, false, false);
  }
  float cb0 = conv_b[ml], cb1 = conv_b[16 + ml];
#pragma unroll
  for (int j = 0; j < 8; ++j) {
    int w = w0 + wq * 16 + j + 8 * hi;
    int base = ((b * HH + h) * WW + w) * COUT;
    y1[base + ml]      = (f16)(acc0[j] + cb0);
    y1[base + 16 + ml] = (f16)(acc1[j] + cb1);
  }
}

// ---------------- K1: GroupNorm stats per (b, group) -------------------------------
__global__ __launch_bounds__(256) void gnstats_kernel(const f16* __restrict__ y1,
                                                      float* __restrict__ stats) {
  int b = blockIdx.x >> 3, g = blockIdx.x & 7;
  int tid = threadIdx.x;
  float s = 0.f, ss = 0.f;
  const f16* base = y1 + (b * HH * WW) * COUT + g * 4;
  for (int pix = tid; pix < HH * WW; pix += 256) {
    const f16* p = base + pix * COUT;
#pragma unroll
    for (int c = 0; c < 4; ++c) { float v = (float)p[c]; s += v; ss += v * v; }
  }
  __shared__ float rs[256], rq[256];
  rs[tid] = s; rq[tid] = ss; __syncthreads();
  for (int st = 128; st > 0; st >>= 1) {
    if (tid < st) { rs[tid] += rs[tid + st]; rq[tid] += rq[tid + st]; }
    __syncthreads();
  }
  if (tid == 0) {
    float inv = 1.f / (HH * WW * 4.f);
    float m = rs[0] * inv, var = rq[0] * inv - m * m;
    stats[(b * 8 + g) * 2] = m;
    stats[(b * 8 + g) * 2 + 1] = rsqrtf(var + 1e-5f);
  }
}

// ---------------- K2a: patch means + coords -> nodes [4096][34] --------------------
__global__ __launch_bounds__(256) void nodes_kernel(const float* __restrict__ x,
                                                    float* __restrict__ nodes) {
  int n = blockIdx.x;
  int b = n >> 8, rem = n & 255, gy = rem >> 4, gx = rem & 15;
  int tid = threadIdx.x, c = tid >> 3, seg = tid & 7;
  const float* p = x + ((b * CIN + c) * HH + gy * 16) * WW + gx * 16;
  float s = 0.f;
#pragma unroll
  for (int rr = 0; rr < 2; ++rr) {
    const float* row = p + (seg * 2 + rr) * WW;
#pragma unroll
    for (int q = 0; q < 16; ++q) s += row[q];
  }
  __shared__ float red[256];
  red[tid] = s; __syncthreads();
  for (int st = 4; st > 0; st >>= 1) {
    if (seg < st) red[tid] += red[tid + st];
    __syncthreads();
  }
  if (seg == 0) nodes[n * 34 + c] = red[tid] * (1.f / 256.f);
  if (tid == 0) { nodes[n * 34 + 32] = gy * (1.f / 16.f); nodes[n * 34 + 33] = gx * (1.f / 16.f); }
}

// ---------------- K2b: h = nodes @ gcn_w (34 -> 32) --------------------------------
__global__ __launch_bounds__(256) void gcnmm_kernel(const float* __restrict__ nodes,
                                                    const float* __restrict__ gcn_w,
                                                    float* __restrict__ hbuf) {
  int idx = blockIdx.x * 256 + threadIdx.x;
  int n = idx >> 5, co = idx & 31;
  float s = 0.f;
#pragma unroll 2
  for (int k = 0; k < 34; ++k) s += nodes[n * 34 + k] * gcn_w[k * 32 + co];
  hbuf[idx] = s;
}

__device__ __forceinline__ float degf_(int gy, int gx) {
  return (float)(1 + (gy > 0) + (gy < 15) + (gx > 0) + (gx < 15));
}

// ---------------- K2c: normalized aggregation + bias + LayerNorm -------------------
__global__ __launch_bounds__(256) void gcnagg_kernel(
    const float* __restrict__ hbuf, const float* __restrict__ gcn_b,
    const float* __restrict__ ln_g, const float* __restrict__ ln_b,
    float* __restrict__ gcn_small) {
  int n = (blockIdx.x * 256 + threadIdx.x) >> 5;   // one node per wave
  int lane = threadIdx.x & 31;                     // lane = channel
  int rem = n & 255, gy = rem >> 4, gx = rem & 15;
  float dinv = rsqrtf(degf_(gy, gx));
  float v = hbuf[n * 32 + lane] * dinv * dinv;     // self loop
  if (gy > 0)  v += hbuf[(n - 16) * 32 + lane] * dinv * rsqrtf(degf_(gy - 1, gx));
  if (gy < 15) v += hbuf[(n + 16) * 32 + lane] * dinv * rsqrtf(degf_(gy + 1, gx));
  if (gx > 0)  v += hbuf[(n - 1)  * 32 + lane] * dinv * rsqrtf(degf_(gy, gx - 1));
  if (gx < 15) v += hbuf[(n + 1)  * 32 + lane] * dinv * rsqrtf(degf_(gy, gx + 1));
  v += gcn_b[lane];
  float s = v;
  for (int off = 16; off > 0; off >>= 1) s += __shfl_xor(s, off, 32);
  float mean = s * (1.f / 32.f);
  float d = v - mean, sq = d * d;
  for (int off = 16; off > 0; off >>= 1) sq += __shfl_xor(sq, off, 32);
  float rstd = rsqrtf(sq * (1.f / 32.f) + 1e-5f);
  // flat store == reference's .view scramble into [B,32,16,16]
  gcn_small[n * 32 + lane] = d * rstd * ln_g[lane] + ln_b[lane];
}

// ---------------- K3: FiLM projection tp = t @ tp_w^T + tp_b -----------------------
__global__ __launch_bounds__(256) void film_kernel(const float* __restrict__ t,
                                                   const float* __restrict__ tp_w,
                                                   const float* __restrict__ tp_b,
                                                   float* __restrict__ tpbuf) {
  int idx = blockIdx.x * 256 + threadIdx.x;
  int b = idx >> 6, j = idx & 63;
  float s = tp_b[j];
  for (int k = 0; k < TD; ++k) s += t[b * TD + k] * tp_w[j * TD + k];
  tpbuf[idx] = s;
}

// ---------------- K4: fused GN+SiLU, bilinear upsample, gate conv (WMMA), merge, FiLM
__global__ __launch_bounds__(128) void fuse_kernel(
    const f16* __restrict__ y1, const float* __restrict__ stats,
    const float* __restrict__ gn_g, const float* __restrict__ gn_b,
    const float* __restrict__ gcn_small, const float* __restrict__ gate_w,
    const float* __restrict__ gate_b, const float* __restrict__ tpbuf,
    float* __restrict__ out) {
  __shared__ __align__(16) f16 slab[3 * 68 * 64];          // [row][col(66 used)][cc]
  __shared__ __align__(16) unsigned char wspace[576 * 32 * 2]; // wlds, later stage
  __shared__ float cscale[32], cbias[32];
  f16*   wlds  = (f16*)wspace;
  float* stage = (float*)wspace;                           // 32*64 floats = 8KB
  const int wt = blockIdx.x, h = blockIdx.y, b = blockIdx.z;
  const int tid = threadIdx.x, w0 = wt * 64;

  if (tid < 32) {
    int g = tid >> 2;
    float m = stats[(b * 8 + g) * 2], rstd = stats[(b * 8 + g) * 2 + 1];
    float sc = rstd * gn_g[tid];
    cscale[tid] = sc; cbias[tid] = gn_b[tid] - m * sc;
  }
  for (int e = tid; e < 576 * 32; e += 128) {
    int co = e / 576, k = e % 576;
    int tap = k / 64, ci = k % 64;
    wlds[co * 576 + k] = (f16)gate_w[(co * 64 + ci) * 9 + tap];
  }
  __syncthreads();

  for (int e = tid; e < 3 * 66 * 64; e += 128) {
    int cc = e & 63, t2 = e >> 6;
    int col = t2 % 66, r = t2 / 66;
    int y = h + r - 1, xg = w0 + col - 1;
    float v = 0.f;
    if ((unsigned)y < 256u && (unsigned)xg < 256u) {
      if (cc < 32) {                                        // conv_feat = SiLU(GN(y1))
        float raw = (float)y1[((b * HH + y) * WW + xg) * COUT + cc];
        float z = raw * cscale[cc] + cbias[cc];
        v = z * sigmoidf_(z);
      } else {                                              // gcn_feat bilinear (x16)
        int co = cc - 32;
        float sy = (y + 0.5f) * (1.f / 16.f) - 0.5f;
        float sx = (xg + 0.5f) * (1.f / 16.f) - 0.5f;
        float fy = floorf(sy), fx = floorf(sx);
        int y0 = (int)fy, x0 = (int)fx;
        float wy = sy - fy, wx = sx - fx;
        int y0c = min(max(y0, 0), 15), y1c = min(max(y0 + 1, 0), 15);
        int x0c = min(max(x0, 0), 15), x1c = min(max(x0 + 1, 0), 15);
        const float* gp = gcn_small + (b * 32 + co) * 256;
        float v00 = gp[y0c * 16 + x0c], v01 = gp[y0c * 16 + x1c];
        float v10 = gp[y1c * 16 + x0c], v11 = gp[y1c * 16 + x1c];
        v = (1.f - wy) * ((1.f - wx) * v00 + wx * v01) +
            wy * ((1.f - wx) * v10 + wx * v11);
      }
    }
    slab[(r * 68 + col) * 64 + cc] = (f16)v;
  }
  __syncthreads();

  const int lane = tid & 31, wq = tid >> 5;
  const int ml = lane & 15, hi = lane >> 4;
  v8f acc0 = {}, acc1 = {};
#pragma unroll
  for (int tap = 0; tap < 9; ++tap) {
    int r = tap / 3, s = tap % 3;
    int col = wq * 16 + ml + s;
#pragma unroll
    for (int cb = 0; cb < 2; ++cb) {
      const f16* ap = &slab[(r * 68 + col) * 64 + cb * 32 + hi * 8];
      v16h A = mk16(*(const h8*)ap, *(const h8*)(ap + 16));
      const f16* bp = &wlds[ml * 576 + tap * 64 + cb * 32 + hi * 8];
      v16h B0 = mk16(*(const h8*)bp,        *(const h8*)(bp + 16));
      v16h B1 = mk16(*(const h8*)(bp + 16 * 576), *(const h8*)(bp + 16 * 576 + 16));
      acc0 = __builtin_amdgcn_wmma_f32_16x16x32_f16(false, A, false, B0, (short)0, acc0, false, false);
      acc1 = __builtin_amdgcn_wmma_f32_16x16x32_f16(false, A, false, B1, (short)0, acc1, false, false);
    }
  }
  __syncthreads();   // wlds -> stage reuse

  float gb0 = gate_b[ml],           gb1 = gate_b[16 + ml];
  float ga0 = tpbuf[b * 64 + ml],   ga1 = tpbuf[b * 64 + 16 + ml];
  float be0 = tpbuf[b * 64 + 32 + ml], be1 = tpbuf[b * 64 + 48 + ml];
#pragma unroll
  for (int j = 0; j < 8; ++j) {
    int m = j + 8 * hi;
    int colc = wq * 16 + m + 1;
    const f16* cp = &slab[(68 + colc) * 64];
    {
      float cf = (float)cp[ml], gf = (float)cp[32 + ml];
      float gate = sigmoidf_(acc0[j] + gb0);
      float merged = gate * cf + (1.f - gate) * gf;
      stage[ml * 64 + wq * 16 + m] = merged * (1.f + ga0) + be0;
    }
    {
      float cf = (float)cp[16 + ml], gf = (float)cp[48 + ml];
      float gate = sigmoidf_(acc1[j] + gb1);
      float merged = gate * cf + (1.f - gate) * gf;
      stage[(16 + ml) * 64 + wq * 16 + m] = merged * (1.f + ga1) + be1;
    }
  }
  __syncthreads();

  int c = tid >> 2, sub = tid & 3;                          // coalesced NCHW writeout
  float* op = out + ((b * COUT + c) * HH + h) * WW + w0 + sub * 16;
  const float* sp = &stage[c * 64 + sub * 16];
#pragma unroll
  for (int q = 0; q < 16; ++q) op[q] = sp[q];
}

extern "C" void kernel_launch(void* const* d_in, const int* in_sizes, int n_in,
                              void* d_out, int out_size, void* d_ws, size_t ws_size,
                              hipStream_t stream) {
  const float* x      = (const float*)d_in[0];
  const float* t      = (const float*)d_in[1];
  const float* conv_w = (const float*)d_in[2];
  const float* conv_b = (const float*)d_in[3];
  const float* gn_g   = (const float*)d_in[4];
  const float* gn_b   = (const float*)d_in[5];
  const float* gcn_w  = (const float*)d_in[6];
  const float* gcn_b  = (const float*)d_in[7];
  const float* ln_g   = (const float*)d_in[8];
  const float* ln_b   = (const float*)d_in[9];
  const float* gate_w = (const float*)d_in[10];
  const float* gate_b = (const float*)d_in[11];
  const float* tp_w   = (const float*)d_in[12];
  const float* tp_b   = (const float*)d_in[13];
  float* out = (float*)d_out;

  char* ws = (char*)d_ws;
  f16*   y1        = (f16*)(ws);                               // 67,108,864 B
  float* stats     = (float*)(ws + 67108864);                  // 1,024 B
  float* nodes     = (float*)(ws + 67109888);                  // 557,056 B
  float* hbuf      = (float*)(ws + 67666944);                  // 524,288 B
  float* gcn_small = (float*)(ws + 68191232);                  // 524,288 B
  float* tpbuf     = (float*)(ws + 68715520);                  // 4,096 B

  conv1_kernel<<<dim3(2, 256, 16), 256, 0, stream>>>(x, conv_w, conv_b, y1);
  gnstats_kernel<<<128, 256, 0, stream>>>(y1, stats);
  nodes_kernel<<<NNODE, 256, 0, stream>>>(x, nodes);
  gcnmm_kernel<<<512, 256, 0, stream>>>(nodes, gcn_w, hbuf);
  gcnagg_kernel<<<512, 256, 0, stream>>>(hbuf, gcn_b, ln_g, ln_b, gcn_small);
  film_kernel<<<4, 256, 0, stream>>>(t, tp_w, tp_b, tpbuf);
  fuse_kernel<<<dim3(4, 256, 16), 128, 0, stream>>>(
      y1, stats, gn_g, gn_b, gcn_small, gate_w, gate_b, tpbuf, out);
}
